// MultiGNN_21337397526759
// MI455X (gfx1250) — matrix operand order repaired and checked by reading, not compiled, not run on hardware
//
#include <hip/hip_runtime.h>
#include <hip/hip_bf16.h>

#define DFEAT 128

typedef __attribute__((ext_vector_type(16))) __bf16 v16bf;
typedef __attribute__((ext_vector_type(8)))  float  v8f;

union FragU { unsigned u[8]; v16bf v; };

__device__ __forceinline__ unsigned short f2bf(float f) {
  unsigned u = __float_as_uint(f);
  u = u + 0x7FFFu + ((u >> 16) & 1u);   // round-to-nearest-even
  return (unsigned short)(u >> 16);
}
__device__ __forceinline__ unsigned pack2(float lo, float hi) {
  return (unsigned)f2bf(lo) | ((unsigned)f2bf(hi) << 16);
}
__device__ __forceinline__ unsigned fkey(float f) {
  unsigned u = __float_as_uint(f);
  return (u & 0x80000000u) ? ~u : (u | 0x80000000u);
}
__device__ __forceinline__ float funkey(unsigned k) {
  return (k & 0x80000000u) ? __uint_as_float(k ^ 0x80000000u) : __uint_as_float(~k);
}

// ---- pack weight [128,128] f32 row-major into B-fragment dword layout -------
// Wp index = ((kt*2 + g)*128 + col)*8 + j ; holds bf16 pair (k0, k0+1),
// k0 = kt*32 + 8*g + 2*j + (j>=4 ? 8 : 0)  (B 32x16 layout: lanes0-15 K={0..7,16..23}, lanes16-31 +8)
__global__ void pack_w_kernel(const float* __restrict__ W, unsigned* __restrict__ Wp) {
  int idx = blockIdx.x * blockDim.x + threadIdx.x;
  if (idx >= 8192) return;
  int j = idx & 7, col = (idx >> 3) & 127, g = (idx >> 10) & 1, kt = idx >> 11;
  int k0 = kt * 32 + 8 * g + 2 * j + (j >= 4 ? 8 : 0);
  Wp[idx] = pack2(W[k0 * DFEAT + col], W[(k0 + 1) * DFEAT + col]);
}

// ---- GEMM: Out[N,128] = A[N,128] @ W[128,128]  (bf16 WMMA, f32 accum) -------
__global__ __launch_bounds__(128) void gemm_wmma_bf16(
    const float* __restrict__ A, const unsigned* __restrict__ Wp,
    float* __restrict__ Out, int nrows) {
  int wid = blockIdx.x * 4 + (threadIdx.x >> 5);
  int lane = threadIdx.x & 31;
  int m0 = wid * 16;
  if (m0 >= nrows) return;                 // wave-uniform: EXEC all-ones inside
  int lm = lane & 15, g = lane >> 4;
  int arowi = m0 + lm; if (arowi >= nrows) arowi = nrows - 1;
  const float* arow = A + (size_t)arowi * DFEAT;
  v8f acc[8] = {};
#pragma unroll
  for (int kt = 0; kt < 4; ++kt) {
    int kbase = kt * 32;
    // A 16x32 bf16 layout: lane<16 K in {0..7,16..23}+kbase ; lane>=16 +8
    const float4* p0 = (const float4*)(arow + kbase + 8 * g);
    const float4* p1 = (const float4*)(arow + kbase + 16 + 8 * g);
    float4 c0 = p0[0], c1 = p0[1], c2 = p1[0], c3 = p1[1];
    FragU a;
    a.u[0] = pack2(c0.x, c0.y); a.u[1] = pack2(c0.z, c0.w);
    a.u[2] = pack2(c1.x, c1.y); a.u[3] = pack2(c1.z, c1.w);
    a.u[4] = pack2(c2.x, c2.y); a.u[5] = pack2(c2.z, c2.w);
    a.u[6] = pack2(c3.x, c3.y); a.u[7] = pack2(c3.z, c3.w);
#pragma unroll
    for (int nt = 0; nt < 8; ++nt) {
      FragU b;
      const unsigned* bp = Wp + (size_t)((kt * 2 + g) * 128 + nt * 16 + lm) * 8;
#pragma unroll
      for (int j = 0; j < 8; ++j) b.u[j] = bp[j];
      acc[nt] = __builtin_amdgcn_wmma_f32_16x16x32_bf16(
          false, a.v, false, b.v, (short)0, acc[nt], false, false);
    }
  }
  // C/D layout: vgpr r -> row m0 + r (+8 for upper lane half), col = nt*16 + lm
  if (m0 + 16 <= nrows) {                  // fast path: no per-store guards
    float* obase = Out + (size_t)(m0 + 8 * g) * DFEAT + lm;
#pragma unroll
    for (int nt = 0; nt < 8; ++nt) {
      float* o = obase + nt * 16;
#pragma unroll
      for (int r = 0; r < 8; ++r) o[(size_t)r * DFEAT] = acc[nt][r];
    }
  } else {                                 // ragged tail (unused for N%16==0)
#pragma unroll
    for (int nt = 0; nt < 8; ++nt) {
      int col = nt * 16 + lm;
#pragma unroll
      for (int r = 0; r < 8; ++r) {
        int m = m0 + r + 8 * g;
        if (m < nrows) Out[(size_t)m * DFEAT + col] = acc[nt][r];
      }
    }
  }
}

// ---- attention dots: a_s[n]=h[n].att_src, a_d[n]=h[n].att_dst ---------------
__global__ void att_dots(const float* __restrict__ H, const float* __restrict__ asv,
                         const float* __restrict__ adv, float* __restrict__ as_o,
                         float* __restrict__ ad_o, int N) {
  int wid = blockIdx.x * (blockDim.x >> 5) + (threadIdx.x >> 5);
  int lane = threadIdx.x & 31;
  if (wid >= N) return;
  const float* row = H + (size_t)wid * DFEAT;
  float s1 = 0.f, s2 = 0.f;
#pragma unroll
  for (int j = 0; j < 4; ++j) {
    int c = lane + 32 * j;
    float v = row[c];
    s1 += v * asv[c]; s2 += v * adv[c];
  }
  for (int off = 16; off; off >>= 1) { s1 += __shfl_down(s1, off); s2 += __shfl_down(s2, off); }
  if (lane == 0) { as_o[wid] = s1; ad_o[wid] = s2; }
}

__global__ void edge_max_kernel(const int* __restrict__ src, const int* __restrict__ dst,
                                const float* __restrict__ as_, const float* __restrict__ ad_,
                                unsigned* __restrict__ mkey, int E) {
  int e = blockIdx.x * blockDim.x + threadIdx.x;
  if (e >= E) return;
  int d = dst[e];
  float l = as_[src[e]] + ad_[d];
  l = l > 0.f ? l : 0.2f * l;
  atomicMax(&mkey[d], fkey(l));
}

__global__ void edge_expsum_kernel(const int* __restrict__ src, const int* __restrict__ dst,
                                   const float* __restrict__ as_, const float* __restrict__ ad_,
                                   const unsigned* __restrict__ mkey, float* __restrict__ z, int E) {
  int e = blockIdx.x * blockDim.x + threadIdx.x;
  if (e >= E) return;
  int d = dst[e];
  float l = as_[src[e]] + ad_[d];
  l = l > 0.f ? l : 0.2f * l;
  atomicAdd(&z[d], __expf(l - funkey(mkey[d])));
}

__global__ void edge_gat_scatter(const int* __restrict__ src, const int* __restrict__ dst,
                                 const float* __restrict__ as_, const float* __restrict__ ad_,
                                 const unsigned* __restrict__ mkey, const float* __restrict__ z,
                                 const float* __restrict__ H, float* __restrict__ out, int E) {
  int wid = blockIdx.x * (blockDim.x >> 5) + (threadIdx.x >> 5);
  int lane = threadIdx.x & 31;
  if (wid >= E) return;
  int s = src[wid], d = dst[wid];
  float l = as_[s] + ad_[d];
  l = l > 0.f ? l : 0.2f * l;
  float attn = __expf(l - funkey(mkey[d])) / (z[d] + 1e-16f);
  float4 v = ((const float4*)(H + (size_t)s * DFEAT))[lane];
  float* o = out + (size_t)d * DFEAT + lane * 4;
  atomicAdd(o + 0, v.x * attn); atomicAdd(o + 1, v.y * attn);
  atomicAdd(o + 2, v.z * attn); atomicAdd(o + 3, v.w * attn);
}

__global__ void edge_gin_scatter(const int* __restrict__ src, const int* __restrict__ dst,
                                 const float* __restrict__ X, float* __restrict__ out, int E) {
  int wid = blockIdx.x * (blockDim.x >> 5) + (threadIdx.x >> 5);
  int lane = threadIdx.x & 31;
  if (wid >= E) return;
  int s = src[wid], d = dst[wid];
  float4 v = ((const float4*)(X + (size_t)s * DFEAT))[lane];
  float* o = out + (size_t)d * DFEAT + lane * 4;
  atomicAdd(o + 0, v.x); atomicAdd(o + 1, v.y);
  atomicAdd(o + 2, v.z); atomicAdd(o + 3, v.w);
}

// ---- per-column batchnorm stats (biased var, eps 1e-5) ----------------------
__global__ void col_stats(const float* __restrict__ X, float* __restrict__ mean,
                          float* __restrict__ rstd, int N) {
  __shared__ float ss[256], sq[256];
  int col = blockIdx.x, t = threadIdx.x;
  float s = 0.f, q = 0.f;
  for (int r = t; r < N; r += 256) {
    float v = X[(size_t)r * DFEAT + col];
    s += v; q += v * v;
  }
  ss[t] = s; sq[t] = q; __syncthreads();
  for (int o = 128; o; o >>= 1) {
    if (t < o) { ss[t] += ss[t + o]; sq[t] += sq[t + o]; }
    __syncthreads();
  }
  if (t == 0) {
    float m = ss[0] / N;
    float v = sq[0] / N - m * m;
    mean[col] = m; rstd[col] = rsqrtf(v + 1e-5f);
  }
}

// ---- y = relu((x-mean)*rstd*g + b); in-place, optional duplicate output -----
__global__ void bn_apply(float* __restrict__ X, float* __restrict__ out2,
                         const float* __restrict__ mean, const float* __restrict__ rstd,
                         const float* __restrict__ gam, const float* __restrict__ bet, int total) {
  int i = blockIdx.x * blockDim.x + threadIdx.x;
  if (i >= total) return;
  int c = i & 127;
  float y = (X[i] - mean[c]) * rstd[c] * gam[c] + bet[c];
  y = fmaxf(y, 0.f);
  X[i] = y;
  if (out2) out2[i] = y;
}

// ---- batched segment-sum pooling of [g1|g2|g3] into conv[16,768] ------------
__global__ void pool384(const float* __restrict__ g1, const float* __restrict__ g2,
                        const float* __restrict__ g3, const int* __restrict__ batch,
                        float* __restrict__ conv, int colbase, int N) {
  int wid = blockIdx.x * (blockDim.x >> 5) + (threadIdx.x >> 5);
  int lane = threadIdx.x & 31;
  if (wid >= N) return;
  int b = batch[wid];
  float* dstp = conv + (size_t)b * 768 + colbase;
  const float* r1 = g1 + (size_t)wid * DFEAT;
  const float* r2 = g2 + (size_t)wid * DFEAT;
  const float* r3 = g3 + (size_t)wid * DFEAT;
#pragma unroll
  for (int j = 0; j < 12; ++j) {
    int c = lane + 32 * j;
    float v = (c < 128) ? r1[c] : ((c < 256) ? r2[c - 128] : r3[c - 256]);
    atomicAdd(dstp + c, v);
  }
}

// ---- tiny FC head: [16,768]@[768,32]+b1 -> @[32,24]+b2 -> relu -> @[24,2]+bc
__global__ void fc_head(const float* __restrict__ conv,
                        const float* __restrict__ W1, const float* __restrict__ b1,
                        const float* __restrict__ W2, const float* __restrict__ b2,
                        const float* __restrict__ Wc, const float* __restrict__ bc,
                        float* __restrict__ out) {
  __shared__ float h1[16 * 32];
  __shared__ float h2[16 * 24];
  int t = threadIdx.x;
  if (t < 512) {
    int i = t >> 5, j = t & 31;
    float s = b1[j];
    for (int k = 0; k < 768; ++k) s += conv[i * 768 + k] * W1[k * 32 + j];
    h1[i * 32 + j] = s;
  }
  __syncthreads();
  if (t < 384) {
    int i = t / 24, j = t % 24;
    float s = b2[j];
    for (int k = 0; k < 32; ++k) s += h1[i * 32 + k] * W2[k * 24 + j];
    h2[i * 24 + j] = fmaxf(s, 0.f);
  }
  __syncthreads();
  if (t < 32) {
    int i = t >> 1, j = t & 1;
    float s = bc[j];
    for (int k = 0; k < 24; ++k) s += h2[i * 24 + k] * Wc[k * 2 + j];
    out[i * 2 + j] = s;
  }
}

// ============================ host side ======================================

struct GinP  { const float *W1, *g1, *be1, *W2, *g2, *be2; };
struct BranchP { const float *W, *asrc, *adst, *bng, *bnb; GinP gin2, gin3; };
struct FcP   { const float *W1, *b1, *W2, *b2, *Wc, *bc; };

static inline const float* FP(void* const* d, int i) { return (const float*)d[i]; }

static GinP mapGin(void* const* d, int base, bool sorted) {
  GinP g;
  if (!sorted) { g.W1 = FP(d, base + 0); g.g1 = FP(d, base + 2); g.be1 = FP(d, base + 3);
                 g.W2 = FP(d, base + 4); g.g2 = FP(d, base + 6); g.be2 = FP(d, base + 7); }
  else         { g.W1 = FP(d, base + 0); g.W2 = FP(d, base + 1); g.be1 = FP(d, base + 4);
                 g.be2 = FP(d, base + 5); g.g1 = FP(d, base + 6); g.g2 = FP(d, base + 7); }
  return g;
}
static BranchP mapBranch(void* const* d, int base, bool sorted) {
  BranchP b;
  if (!sorted) { b.W = FP(d, base + 0); b.asrc = FP(d, base + 1); b.adst = FP(d, base + 2);
                 b.bng = FP(d, base + 4); b.bnb = FP(d, base + 5);
                 b.gin2 = mapGin(d, base + 6, false); b.gin3 = mapGin(d, base + 14, false); }
  else         { b.bnb = FP(d, base + 0); b.bng = FP(d, base + 1); b.W = FP(d, base + 2);
                 b.asrc = FP(d, base + 3); b.adst = FP(d, base + 4);
                 b.gin2 = mapGin(d, base + 6, true); b.gin3 = mapGin(d, base + 14, true); }
  return b;
}
static FcP mapFc(void* const* d, int base, bool sorted) {
  FcP f;
  if (!sorted) { f.W1 = FP(d, base + 0); f.b1 = FP(d, base + 1); f.W2 = FP(d, base + 2);
                 f.b2 = FP(d, base + 3); f.Wc = FP(d, base + 4); f.bc = FP(d, base + 5); }
  else         { f.W1 = FP(d, base + 0); f.W2 = FP(d, base + 1); f.Wc = FP(d, base + 2);
                 f.b1 = FP(d, base + 3); f.b2 = FP(d, base + 4); f.bc = FP(d, base + 5); }
  return f;
}

struct Ws {
  float *H, *G1, *G2, *T;
  float *a_s, *a_d, *z, *mean, *rstd, *conv;
  unsigned *mkey, *Wp;
};

static void run_branch(const float* x, const int* src, const int* dst, const BranchP& P,
                       const int* batch, int colbase, int N, int E, const Ws& w,
                       hipStream_t st) {
  int nd = N * DFEAT;
  int gemmBlocks = ((N + 15) / 16 + 3) / 4;
  int nodeW = (N + 7) / 8, edgeT = (E + 255) / 256, edgeW = (E + 7) / 8;
  int elemB = (nd + 255) / 256;

  hipMemsetAsync(w.mkey, 0, (size_t)N * 4, st);
  hipMemsetAsync(w.z, 0, (size_t)N * 4, st);
  hipMemsetAsync(w.G1, 0, (size_t)nd * 4, st);

  // ---- GAT layer ----
  pack_w_kernel<<<32, 256, 0, st>>>(P.W, w.Wp);
  gemm_wmma_bf16<<<gemmBlocks, 128, 0, st>>>(x, w.Wp, w.H, N);
  att_dots<<<nodeW, 256, 0, st>>>(w.H, P.asrc, P.adst, w.a_s, w.a_d, N);
  edge_max_kernel<<<edgeT, 256, 0, st>>>(src, dst, w.a_s, w.a_d, w.mkey, E);
  edge_expsum_kernel<<<edgeT, 256, 0, st>>>(src, dst, w.a_s, w.a_d, w.mkey, w.z, E);
  edge_gat_scatter<<<edgeW, 256, 0, st>>>(src, dst, w.a_s, w.a_d, w.mkey, w.z, w.H, w.G1, E);
  col_stats<<<128, 256, 0, st>>>(w.G1, w.mean, w.rstd, N);
  bn_apply<<<elemB, 256, 0, st>>>(w.G1, w.H, w.mean, w.rstd, P.bng, P.bnb, nd); // g1; H=g1 copy

  // ---- GIN layer 2: H = g1 + agg(g1) ----
  edge_gin_scatter<<<edgeW, 256, 0, st>>>(src, dst, w.G1, w.H, E);
  pack_w_kernel<<<32, 256, 0, st>>>(P.gin2.W1, w.Wp);
  gemm_wmma_bf16<<<gemmBlocks, 128, 0, st>>>(w.H, w.Wp, w.T, N);
  col_stats<<<128, 256, 0, st>>>(w.T, w.mean, w.rstd, N);
  bn_apply<<<elemB, 256, 0, st>>>(w.T, nullptr, w.mean, w.rstd, P.gin2.g1, P.gin2.be1, nd);
  pack_w_kernel<<<32, 256, 0, st>>>(P.gin2.W2, w.Wp);
  gemm_wmma_bf16<<<gemmBlocks, 128, 0, st>>>(w.T, w.Wp, w.H, N);
  col_stats<<<128, 256, 0, st>>>(w.H, w.mean, w.rstd, N);
  bn_apply<<<elemB, 256, 0, st>>>(w.H, w.G2, w.mean, w.rstd, P.gin2.g2, P.gin2.be2, nd); // g2; H=g2 copy

  // ---- GIN layer 3: H = g2 + agg(g2) ----
  edge_gin_scatter<<<edgeW, 256, 0, st>>>(src, dst, w.G2, w.H, E);
  pack_w_kernel<<<32, 256, 0, st>>>(P.gin3.W1, w.Wp);
  gemm_wmma_bf16<<<gemmBlocks, 128, 0, st>>>(w.H, w.Wp, w.T, N);
  col_stats<<<128, 256, 0, st>>>(w.T, w.mean, w.rstd, N);
  bn_apply<<<elemB, 256, 0, st>>>(w.T, nullptr, w.mean, w.rstd, P.gin3.g1, P.gin3.be1, nd);
  pack_w_kernel<<<32, 256, 0, st>>>(P.gin3.W2, w.Wp);
  gemm_wmma_bf16<<<gemmBlocks, 128, 0, st>>>(w.T, w.Wp, w.H, N);
  col_stats<<<128, 256, 0, st>>>(w.H, w.mean, w.rstd, N);
  bn_apply<<<elemB, 256, 0, st>>>(w.H, nullptr, w.mean, w.rstd, P.gin3.g2, P.gin3.be2, nd); // g3 = H

  // ---- pooling ----
  pool384<<<nodeW, 256, 0, st>>>(w.G1, w.G2, w.H, batch, w.conv, colbase, N);
}

extern "C" void kernel_launch(void* const* d_in, const int* in_sizes, int n_in,
                              void* d_out, int out_size, void* d_ws, size_t ws_size,
                              hipStream_t stream) {
  if (n_in < 55) return;

  const float *x_r, *x_p;
  const int *edge_r, *edge_p, *batch;
  BranchP br, bp;
  FcP fc;
  int N, E;

  if (in_sizes[0] < 1000000) {
    // fully alphabetical (jax pytree) flattening:
    // batch, edge_p, edge_r, params{fc,p,r}, x_p, x_r
    batch = (const int*)d_in[0];
    edge_p = (const int*)d_in[1];
    edge_r = (const int*)d_in[2];
    fc = mapFc(d_in, 3, true);
    bp = mapBranch(d_in, 9, true);
    br = mapBranch(d_in, 31, true);
    x_p = FP(d_in, 53);
    x_r = FP(d_in, 54);
    N = in_sizes[54] / DFEAT;
    E = in_sizes[1] / 2;
  } else {
    // top-level insertion order: x_r, edge_r, x_p, edge_p, batch, params...
    x_r = FP(d_in, 0);
    edge_r = (const int*)d_in[1];
    x_p = FP(d_in, 2);
    edge_p = (const int*)d_in[3];
    batch = (const int*)d_in[4];
    N = in_sizes[0] / DFEAT;
    E = in_sizes[1] / 2;
    if (in_sizes[5] == 24576) {           // nested alphabetical: fc first
      fc = mapFc(d_in, 5, true);
      bp = mapBranch(d_in, 11, true);
      br = mapBranch(d_in, 33, true);
    } else {                              // full insertion order
      br = mapBranch(d_in, 5, false);
      bp = mapBranch(d_in, 27, false);
      fc = mapFc(d_in, 49, false);
    }
  }

  // ---- carve workspace ----
  size_t nd4 = ((size_t)N * DFEAT * 4 + 255) & ~(size_t)255;
  size_t n4  = ((size_t)N * 4 + 255) & ~(size_t)255;
  size_t need = 4 * nd4 + 4 * n4 + 3 * 512 + 8192 * 4 + 16 * 768 * 4 + 4096;
  if (ws_size < need) return;

  char* p = (char*)d_ws;
  Ws w;
  w.H  = (float*)p; p += nd4;
  w.G1 = (float*)p; p += nd4;
  w.G2 = (float*)p; p += nd4;
  w.T  = (float*)p; p += nd4;
  w.a_s = (float*)p; p += n4;
  w.a_d = (float*)p; p += n4;
  w.z   = (float*)p; p += n4;
  w.mkey = (unsigned*)p; p += n4;
  w.mean = (float*)p; p += 512;
  w.rstd = (float*)p; p += 512;
  w.Wp   = (unsigned*)p; p += 8192 * 4;
  w.conv = (float*)p; p += 16 * 768 * 4;

  hipMemsetAsync(w.conv, 0, 16 * 768 * 4, stream);

  const int* src_r = edge_r;       const int* dst_r = edge_r + E;
  const int* src_p = edge_p;       const int* dst_p = edge_p + E;

  run_branch(x_r, src_r, dst_r, br, batch, 0,   N, E, w, stream);
  run_branch(x_p, src_p, dst_p, bp, batch, 384, N, E, w, stream);

  fc_head<<<1, 512, 0, stream>>>(w.conv, fc.W1, fc.b1, fc.W2, fc.b2, fc.Wc, fc.bc,
                                 (float*)d_out);
}
